// uniform_loss_85590108274881
// MI455X (gfx1250) — matrix operand order repaired and checked by reading, not compiled, not run on hardware
//
#include <hip/hip_runtime.h>
#include <hip/hip_bf16.h>

typedef __bf16 bf16_t;
typedef __attribute__((ext_vector_type(16))) __bf16 v16bf;
typedef __attribute__((ext_vector_type(8)))  __bf16 v8bf;
typedef __attribute__((ext_vector_type(8)))  float  v8f;

constexpr int   kN       = 12288;
constexpr int   kD       = 128;
constexpr int   kCChunks = 8;                    // column split for parallelism
constexpr float kScale   = 20.60992915555662f;   // log2(e) / 0.07
constexpr float kLn2     = 0.6931471805599453f;

// ---------------- kernel 0: zero the row-sum accumulator ----------------
__global__ void ul_zero(float* __restrict__ S, int n) {
  int i = blockIdx.x * blockDim.x + threadIdx.x;
  if (i < n) S[i] = 0.0f;
}

// ---------------- kernel 1: fp32 -> bf16 hi/lo split ----------------
__global__ void ul_split(const float* __restrict__ x,
                         bf16_t* __restrict__ xh,
                         bf16_t* __restrict__ xl, int n) {
  int i = blockIdx.x * blockDim.x + threadIdx.x;
  if (i < n) {
    float v = x[i];
    bf16_t h = (bf16_t)v;           // RNE to bf16
    xh[i] = h;
    xl[i] = (bf16_t)(v - (float)h); // residual: next 8 mantissa bits
  }
}

// ---------------- kernel 2: WMMA Gram tile + exp + row-sum ----------------
// Grid: (kN/128, kCChunks), block: 256 threads = 8 waves (wave32).
// Wave w owns rows [blockIdx.x*128 + w*16, +16). A fragments (all 4 k-steps,
// hi+lo) are hoisted into registers; B pointers advance by an immediate-
// addressable stride so every load is base+imm. Row partial sums accumulate
// across all panels; one shuffle-reduce + 8 atomics per wave at the end.
__global__ __launch_bounds__(256) void ul_gemm(const bf16_t* __restrict__ xh,
                                               const bf16_t* __restrict__ xl,
                                               float* __restrict__ S) {
  const int lane = threadIdx.x & 31;
  const int wave = threadIdx.x >> 5;
  const int half = lane >> 4;       // K sub-range select per ISA fragment layout
  const int l16  = lane & 15;

  const int ri   = blockIdx.x * 128 + wave * 16;
  const int cPer = kN / kCChunks;
  const int cBeg = blockIdx.y * cPer;

  // ---- A fragments: 16x32 bf16 per k-step; lane half h holds
  //      K = k0+h*16..+15 (contiguous). Loop-invariant.
  const size_t aBase = (size_t)(ri + l16) * kD + half * 16;
  v16bf ah[4], al[4];
#pragma unroll
  for (int kk = 0; kk < 4; ++kk) {
    const bf16_t* pah = xh + aBase + kk * 32;
    const bf16_t* pal = xl + aBase + kk * 32;
    v8bf h0 = *(const v8bf*)(pah);
    v8bf h1 = *(const v8bf*)(pah + 8);
    v8bf l0 = *(const v8bf*)(pal);
    v8bf l1 = *(const v8bf*)(pal + 8);
    ah[kk] = __builtin_shufflevector(h0, h1, 0,1,2,3,4,5,6,7,8,9,10,11,12,13,14,15);
    al[kk] = __builtin_shufflevector(l0, l1, 0,1,2,3,4,5,6,7,8,9,10,11,12,13,14,15);
  }

  // ---- B base pointers: column n = row n of x (Gram symmetry); lane half h
  //      holds K = k0+h*16..+15 -> contiguous 32B. Tile/k offsets are
  //      compile-time immediates; pointers bump once per 128-col panel.
  const bf16_t* pbh = xh + (size_t)(cBeg + l16) * kD + half * 16;
  const bf16_t* pbl = xl + (size_t)(cBeg + l16) * kD + half * 16;
  constexpr int kPanelStep = 128 * kD;

  float rs[8];                       // per-row exp partial sums (persist)
#pragma unroll
  for (int r = 0; r < 8; ++r) rs[r] = 0.0f;

  for (int cj = cBeg; cj < cBeg + cPer; cj += 128) {
    // prefetch next panel's B rows (L2-resident; hides latency)
    __builtin_prefetch(pbh + kPanelStep, 0, 3);
    __builtin_prefetch(pbl + kPanelStep, 0, 3);

    v8f acc[8];
#pragma unroll
    for (int t = 0; t < 8; ++t) acc[t] = v8f{};

#pragma unroll
    for (int kk = 0; kk < 4; ++kk) {
#pragma unroll
      for (int t = 0; t < 8; ++t) {
        const int off = t * 16 * kD + kk * 32;   // immediate per unrolled iter
        v16bf bh = *(const v16bf*)(pbh + off);
        v16bf bl = *(const v16bf*)(pbl + off);
        acc[t] = __builtin_amdgcn_wmma_f32_16x16x32_bf16(
            false, ah[kk], false, bh, (short)0, acc[t], false, false);
        acc[t] = __builtin_amdgcn_wmma_f32_16x16x32_bf16(
            false, ah[kk], false, bl, (short)0, acc[t], false, false);
        acc[t] = __builtin_amdgcn_wmma_f32_16x16x32_bf16(
            false, al[kk], false, bh, (short)0, acc[t], false, false);
      }
    }

    // sim -> exp(sim/T) = 2^(sim * log2e/T); accumulate per-row sums.
    // C layout: acc[t][r] = sim[ri + r + 8*half][cj + t*16 + l16].
#pragma unroll
    for (int t = 0; t < 8; ++t) {
#pragma unroll
      for (int r = 0; r < 8; ++r) {
        rs[r] += __builtin_amdgcn_exp2f(acc[t][r] * kScale);
      }
    }

    pbh += kPanelStep;
    pbl += kPanelStep;
  }

  // One reduction + atomic per row at the very end (8 atomics per wave).
#pragma unroll
  for (int r = 0; r < 8; ++r) {
    float v = rs[r];
    v += __shfl_xor(v, 1, 16);       // reduce within 16-lane half (same row)
    v += __shfl_xor(v, 2, 16);
    v += __shfl_xor(v, 4, 16);
    v += __shfl_xor(v, 8, 16);
    if (l16 == 0) atomicAdd(&S[ri + r + half * 8], v);
  }
}

// ---------------- kernel 3: mean(log(S)) ----------------
__global__ void ul_finish(const float* __restrict__ S, float* __restrict__ out, int n) {
  __shared__ float sm[256];
  float a = 0.0f;
  for (int i = threadIdx.x; i < n; i += 256)
    a += __builtin_amdgcn_logf(S[i]);          // log2; scale by ln2 once
  sm[threadIdx.x] = a;
  __syncthreads();
  for (int s = 128; s > 0; s >>= 1) {
    if ((int)threadIdx.x < s) sm[threadIdx.x] += sm[threadIdx.x + s];
    __syncthreads();
  }
  if (threadIdx.x == 0) out[0] = sm[0] * (kLn2 / (float)n);
}

extern "C" void kernel_launch(void* const* d_in, const int* in_sizes, int n_in,
                              void* d_out, int out_size, void* d_ws, size_t ws_size,
                              hipStream_t stream) {
  (void)in_sizes; (void)n_in; (void)out_size; (void)ws_size;
  const float* x = (const float*)d_in[0];
  float* out = (float*)d_out;

  char* ws = (char*)d_ws;
  bf16_t* xh = (bf16_t*)ws;
  bf16_t* xl = (bf16_t*)(ws + (size_t)kN * kD * sizeof(bf16_t));
  float*  S  = (float*) (ws + 2 * (size_t)kN * kD * sizeof(bf16_t));

  ul_zero <<<(kN + 255) / 256, 256, 0, stream>>>(S, kN);
  ul_split<<<(kN * kD + 255) / 256, 256, 0, stream>>>(x, xh, xl, kN * kD);
  dim3 grid(kN / 128, kCChunks);
  ul_gemm <<<grid, 256, 0, stream>>>(xh, xl, S);
  ul_finish<<<1, 256, 0, stream>>>(S, out, kN);
}
